// MultiHeadAttention_34892314313486
// MI455X (gfx1250) — matrix operand order repaired and checked by reading, not compiled, not run on hardware
//
#include <hip/hip_runtime.h>
#include <hip/hip_bf16.h>

typedef __attribute__((ext_vector_type(16))) _Float16 v16h;
typedef __attribute__((ext_vector_type(8)))  float    v8f;

#define DIM_IN 1024
#define NHEAD  16
#define DQ     64
#define SEQ    2048
#define BATCH  4

union F16x16 {
    v16h v;
    unsigned u[8];
    _Float16 h[16];
};

// 16-bit A-matrix 16x32 layout (ISA 7.12.2): lane l holds row m=l%16,
// VGPR i holds K pair starting at (i/4)*16 + (l/16)*8 + (i%4)*2.
__device__ __forceinline__ int k2A(int i, int hh) {
    return ((i >> 2) << 4) + (hh << 3) + ((i & 3) << 1);
}
// 16-bit B-matrix 32x16 layout: lane l holds col n=l%16,
// VGPR i holds K pair starting at (i/4)*16 + (l/16)*8 + (i%4)*2.
__device__ __forceinline__ int k2B(int i, int hh) {
    return ((i >> 2) << 4) + (hh << 3) + ((i & 3) << 1);
}

// ---------------------------------------------------------------------------
// Kernel 0: convert + transpose weights to f16.
//  Wq/Wk/Wv [H][D][64] f32 -> WqT/WkT/WvT [H][64][D] f16
//  Wo [D][1024] f32       -> WoT [1024][D] f16
// ---------------------------------------------------------------------------
__global__ void __launch_bounds__(256)
convert_weights_kernel(const float* __restrict__ Wq, const float* __restrict__ Wk,
                       const float* __restrict__ Wv, const float* __restrict__ Wo,
                       _Float16* __restrict__ WqT, _Float16* __restrict__ WkT,
                       _Float16* __restrict__ WvT, _Float16* __restrict__ WoT)
{
    const int PERW = NHEAD * DQ * DIM_IN;  // 1048576
    int gid = blockIdx.x * blockDim.x + threadIdx.x;
    if (gid < 3 * PERW) {
        int which = gid / PERW;
        int r     = gid % PERW;
        int h  = r / (DQ * DIM_IN);
        int rr = r % (DQ * DIM_IN);
        int n  = rr / DIM_IN;
        int d  = rr % DIM_IN;
        const float* W = (which == 0) ? Wq : (which == 1) ? Wk : Wv;
        _Float16*    T = (which == 0) ? WqT : (which == 1) ? WkT : WvT;
        T[r] = (_Float16)W[(size_t)h * DIM_IN * DQ + (size_t)d * DQ + n];
    } else {
        int r    = gid - 3 * PERW;         // 0 .. 1048575
        int nout = r / DIM_IN;
        int k    = r % DIM_IN;
        WoT[r] = (_Float16)Wo[(size_t)k * DIM_IN + nout];
    }
}

// ---------------------------------------------------------------------------
// Kernel 1: QKV projection.  One wave computes a 16(s) x 64(dq) strip for
// one (b, h, which). which: 0=Q->qh, 1=K->kh (both [bh][s][64] f16),
// 2=V->vT ([bh][64][S] f16, transposed for PV B-operand).
// ---------------------------------------------------------------------------
__global__ void __launch_bounds__(128)
qkv_proj_kernel(const float* __restrict__ Q, const float* __restrict__ K,
                const float* __restrict__ V,
                const _Float16* __restrict__ WqT, const _Float16* __restrict__ WkT,
                const _Float16* __restrict__ WvT,
                const float* __restrict__ bq, const float* __restrict__ bk,
                const float* __restrict__ bv,
                _Float16* __restrict__ qh, _Float16* __restrict__ kh,
                _Float16* __restrict__ vT)
{
    const int lane = threadIdx.x & 31;
    const int wave = threadIdx.x >> 5;
    const int n16  = lane & 15;
    const int hh   = lane >> 4;
    const int s0   = (blockIdx.x * 4 + wave) * 16;
    const int bh   = blockIdx.y;     // b*NHEAD + h
    const int b    = bh >> 4;
    const int h    = bh & 15;
    const int which = blockIdx.z;

    const float*    X    = (which == 0) ? Q  : (which == 1) ? K  : V;
    const _Float16* WT   = (which == 0) ? WqT : (which == 1) ? WkT : WvT;
    const float*    bias = (which == 0) ? bq : (which == 1) ? bk : bv;

    v8f acc[4];
    #pragma unroll
    for (int f = 0; f < 4; ++f)
        #pragma unroll
        for (int r = 0; r < 8; ++r) acc[f][r] = 0.0f;

    const float*    arow  = X + ((size_t)b * SEQ + (s0 + n16)) * DIM_IN;
    const _Float16* wbase = WT + (size_t)h * DQ * DIM_IN;

    for (int d0 = 0; d0 < DIM_IN; d0 += 32) {
        F16x16 a;
        #pragma unroll
        for (int i = 0; i < 8; ++i) {
            float2 x = *reinterpret_cast<const float2*>(arow + d0 + k2A(i, hh));
            a.h[2 * i]     = (_Float16)x.x;
            a.h[2 * i + 1] = (_Float16)x.y;
        }
        #pragma unroll
        for (int f = 0; f < 4; ++f) {
            F16x16 bf;
            const _Float16* wcol = wbase + (size_t)(f * 16 + n16) * DIM_IN + d0;
            #pragma unroll
            for (int i = 0; i < 8; ++i)
                bf.u[i] = *reinterpret_cast<const unsigned*>(wcol + k2B(i, hh));
            acc[f] = __builtin_amdgcn_wmma_f32_16x16x32_f16(
                false, a.v, false, bf.v, (short)0, acc[f], false, false);
        }
    }

    if (which < 2) {
        _Float16* dst = ((which == 0) ? qh : kh) + (size_t)bh * SEQ * DQ;
        #pragma unroll
        for (int f = 0; f < 4; ++f) {
            float bb = bias[h * DQ + f * 16 + n16];
            #pragma unroll
            for (int r = 0; r < 8; ++r) {
                int m = hh * 8 + r;
                dst[(size_t)(s0 + m) * DQ + f * 16 + n16] = (_Float16)(acc[f][r] + bb);
            }
        }
    } else {
        _Float16* dst = vT + (size_t)bh * DQ * SEQ;
        #pragma unroll
        for (int f = 0; f < 4; ++f) {
            float bb = bias[h * DQ + f * 16 + n16];
            #pragma unroll
            for (int r = 0; r < 8; ++r) {
                int m = hh * 8 + r;
                dst[(size_t)(f * 16 + n16) * SEQ + (s0 + m)] = (_Float16)(acc[f][r] + bb);
            }
        }
    }
}

// ---------------------------------------------------------------------------
// Kernel 2: flash attention. One wave owns a 16-row q tile for one (b,h);
// streams T in steps of 64 (4 score tiles per online-softmax update).
// Row-sums are computed with an extra WMMA against an all-ones B fragment
// (exact: any row-consistent max m~ works in softmax, and l then sums the
// same f16-rounded P that feeds PxV). Per 64 cols: 18 WMMAs, ~32 shuffles.
// ---------------------------------------------------------------------------
__global__ void __launch_bounds__(128)
attn_kernel(const _Float16* __restrict__ qh, const _Float16* __restrict__ kh,
            const _Float16* __restrict__ vT, _Float16* __restrict__ concat)
{
    __shared__ _Float16 pbuf[4][16 * 64];

    const int lane = threadIdx.x & 31;
    const int wave = threadIdx.x >> 5;
    const int n16  = lane & 15;
    const int hh   = lane >> 4;
    const int s0   = (blockIdx.x * 4 + wave) * 16;
    const int bh   = blockIdx.y;
    const int b    = bh >> 4;
    const int h    = bh & 15;

    const _Float16* qbase = qh + (size_t)bh * SEQ * DQ;
    const _Float16* kbase = kh + (size_t)bh * SEQ * DQ;
    const _Float16* vbase = vT + (size_t)bh * DQ * SEQ;

    // q A-fragments (dq = 64 -> 2 fragments of 16x32), held for whole loop
    F16x16 qf[2];
    {
        const _Float16* qrow = qbase + (size_t)(s0 + n16) * DQ;
        #pragma unroll
        for (int kk = 0; kk < 2; ++kk)
            #pragma unroll
            for (int i = 0; i < 8; ++i)
                qf[kk].u[i] = *reinterpret_cast<const unsigned*>(qrow + kk * 32 + k2A(i, hh));
    }

    // all-ones B fragment for row-sum WMMA
    F16x16 ones;
    #pragma unroll
    for (int i = 0; i < 16; ++i) ones.h[i] = (_Float16)1.0f;

    v8f acc[4];
    #pragma unroll
    for (int f = 0; f < 4; ++f)
        #pragma unroll
        for (int r = 0; r < 8; ++r) acc[f][r] = 0.0f;
    float mi[8], li[8];
    #pragma unroll
    for (int r = 0; r < 8; ++r) { mi[r] = -1e30f; li[r] = 0.0f; }
    const float scale = 0.125f;   // 1/sqrt(64)

    _Float16* myp = pbuf[wave];

    for (int t0 = 0; t0 < SEQ; t0 += 64) {
        // ---- scores: four 16x16 tiles, dq=64 contraction (2 WMMAs each) ----
        v8f sc[4];
        #pragma unroll
        for (int tt = 0; tt < 4; ++tt) {
            v8f c;
            #pragma unroll
            for (int r = 0; r < 8; ++r) c[r] = 0.0f;
            const _Float16* krow = kbase + (size_t)(t0 + tt * 16 + n16) * DQ;
            #pragma unroll
            for (int kk = 0; kk < 2; ++kk) {
                F16x16 bf;
                #pragma unroll
                for (int i = 0; i < 8; ++i)
                    bf.u[i] = *reinterpret_cast<const unsigned*>(krow + kk * 32 + k2B(i, hh));
                c = __builtin_amdgcn_wmma_f32_16x16x32_f16(
                    false, qf[kk].v, false, bf.v, (short)0, c, false, false);
            }
            sc[tt] = c;
        }

        // ---- online softmax update over these 64 columns ----
        float alpha[8];
        #pragma unroll
        for (int r = 0; r < 8; ++r) {
            float mx = fmaxf(fmaxf(sc[0][r], sc[1][r]), fmaxf(sc[2][r], sc[3][r]));
            #pragma unroll
            for (int msk = 1; msk < 16; msk <<= 1)
                mx = fmaxf(mx, __shfl_xor(mx, msk, 32));
            float mn = fmaxf(mi[r], mx * scale);
            alpha[r] = __expf(mi[r] - mn);
            mi[r] = mn;
            // p = exp(s*scale - mn), converted to f16 and staged to LDS
            const int m = hh * 8 + r;
            #pragma unroll
            for (int tt = 0; tt < 4; ++tt) {
                float p = __expf(sc[tt][r] * scale - mn);
                myp[m * 64 + tt * 16 + n16] = (_Float16)p;
            }
        }
        #pragma unroll
        for (int f = 0; f < 4; ++f)
            #pragma unroll
            for (int r = 0; r < 8; ++r)
                acc[f][r] *= alpha[r];
        __syncthreads();

        // ---- P A-fragments (16x64 -> 2 fragments) from LDS ----
        F16x16 pf[2];
        {
            const _Float16* prow = myp + n16 * 64;
            #pragma unroll
            for (int kk = 0; kk < 2; ++kk)
                #pragma unroll
                for (int i = 0; i < 8; ++i)
                    pf[kk].u[i] = *reinterpret_cast<const unsigned*>(prow + kk * 32 + k2A(i, hh));
        }

        // ---- row sums of P via WMMA with ones (replaces shuffle tree) ----
        v8f rs;
        #pragma unroll
        for (int r = 0; r < 8; ++r) rs[r] = 0.0f;
        rs = __builtin_amdgcn_wmma_f32_16x16x32_f16(
            false, pf[0].v, false, ones.v, (short)0, rs, false, false);
        rs = __builtin_amdgcn_wmma_f32_16x16x32_f16(
            false, pf[1].v, false, ones.v, (short)0, rs, false, false);
        #pragma unroll
        for (int r = 0; r < 8; ++r)
            li[r] = li[r] * alpha[r] + rs[r];

        // ---- acc += P x V (4 output-column fragments, K=64) ----
        #pragma unroll
        for (int f = 0; f < 4; ++f) {
            const _Float16* vrow = vbase + (size_t)(f * 16 + n16) * SEQ + t0;
            #pragma unroll
            for (int kk = 0; kk < 2; ++kk) {
                F16x16 bf;
                #pragma unroll
                for (int i = 0; i < 8; ++i)
                    bf.u[i] = *reinterpret_cast<const unsigned*>(vrow + kk * 32 + k2B(i, hh));
                acc[f] = __builtin_amdgcn_wmma_f32_16x16x32_f16(
                    false, pf[kk].v, false, bf.v, (short)0, acc[f], false, false);
            }
        }
        __syncthreads();
    }

    // ---- normalize and write concat[b][s][h*64 + dv] ----
    _Float16* dst = concat + (size_t)b * SEQ * (NHEAD * DQ) + (size_t)h * DQ;
    #pragma unroll
    for (int f = 0; f < 4; ++f) {
        #pragma unroll
        for (int r = 0; r < 8; ++r) {
            int m = hh * 8 + r;
            float inv = 1.0f / li[r];
            dst[(size_t)(s0 + m) * (NHEAD * DQ) + f * 16 + n16] =
                (_Float16)(acc[f][r] * inv);
        }
    }
}

// ---------------------------------------------------------------------------
// Kernel 3: output projection. One wave: 16(s) x 64(nout) strip, K=1024.
// ---------------------------------------------------------------------------
__global__ void __launch_bounds__(128)
out_proj_kernel(const _Float16* __restrict__ concat, const _Float16* __restrict__ WoT,
                const float* __restrict__ bo, float* __restrict__ out)
{
    const int lane = threadIdx.x & 31;
    const int wave = threadIdx.x >> 5;
    const int n16  = lane & 15;
    const int hh   = lane >> 4;
    const int ng   = blockIdx.x * 4 + wave;   // 0..15
    const int n0   = ng * 64;
    const int s0   = blockIdx.y * 16;
    const int b    = blockIdx.z;

    v8f acc[4];
    #pragma unroll
    for (int f = 0; f < 4; ++f)
        #pragma unroll
        for (int r = 0; r < 8; ++r) acc[f][r] = 0.0f;

    const _Float16* arow = concat + ((size_t)b * SEQ + s0 + n16) * DIM_IN;

    for (int k0 = 0; k0 < DIM_IN; k0 += 32) {
        F16x16 a;
        #pragma unroll
        for (int i = 0; i < 8; ++i)
            a.u[i] = *reinterpret_cast<const unsigned*>(arow + k0 + k2A(i, hh));
        #pragma unroll
        for (int f = 0; f < 4; ++f) {
            F16x16 bf;
            const _Float16* wrow = WoT + (size_t)(n0 + f * 16 + n16) * DIM_IN + k0;
            #pragma unroll
            for (int i = 0; i < 8; ++i)
                bf.u[i] = *reinterpret_cast<const unsigned*>(wrow + k2B(i, hh));
            acc[f] = __builtin_amdgcn_wmma_f32_16x16x32_f16(
                false, a.v, false, bf.v, (short)0, acc[f], false, false);
        }
    }

    float* dst = out + (size_t)b * SEQ * DIM_IN;
    #pragma unroll
    for (int f = 0; f < 4; ++f) {
        float bb = bo[n0 + f * 16 + n16];
        #pragma unroll
        for (int r = 0; r < 8; ++r) {
            int m = hh * 8 + r;
            dst[(size_t)(s0 + m) * DIM_IN + n0 + f * 16 + n16] = acc[f][r] + bb;
        }
    }
}

// ---------------------------------------------------------------------------
extern "C" void kernel_launch(void* const* d_in, const int* in_sizes, int n_in,
                              void* d_out, int out_size, void* d_ws, size_t ws_size,
                              hipStream_t stream)
{
    (void)in_sizes; (void)n_in; (void)out_size; (void)ws_size;

    const float* Q  = (const float*)d_in[0];
    const float* K  = (const float*)d_in[1];
    const float* V  = (const float*)d_in[2];
    const float* Wq = (const float*)d_in[3];
    const float* bq = (const float*)d_in[4];
    const float* Wk = (const float*)d_in[5];
    const float* bk = (const float*)d_in[6];
    const float* Wv = (const float*)d_in[7];
    const float* bv = (const float*)d_in[8];
    const float* Wo = (const float*)d_in[9];
    const float* bo = (const float*)d_in[10];
    float* out = (float*)d_out;

    char* ws = (char*)d_ws;
    size_t off = 0;
    auto carve = [&](size_t bytes) -> void* {
        void* p = ws + off;
        off += (bytes + 255) & ~(size_t)255;
        return p;
    };
    const size_t WBYTES = (size_t)NHEAD * DQ * DIM_IN * sizeof(_Float16);       // 2 MB
    const size_t PBYTES = (size_t)BATCH * NHEAD * SEQ * DQ * sizeof(_Float16);  // 16 MB
    _Float16* WqT    = (_Float16*)carve(WBYTES);
    _Float16* WkT    = (_Float16*)carve(WBYTES);
    _Float16* WvT    = (_Float16*)carve(WBYTES);
    _Float16* WoT    = (_Float16*)carve((size_t)DIM_IN * DIM_IN * sizeof(_Float16));
    _Float16* qhBuf  = (_Float16*)carve(PBYTES);
    _Float16* khBuf  = (_Float16*)carve(PBYTES);
    _Float16* vTBuf  = (_Float16*)carve(PBYTES);
    _Float16* concat = (_Float16*)carve((size_t)BATCH * SEQ * DIM_IN * sizeof(_Float16));

    // 0) weights -> f16, transposed for contiguous B-operand pairs
    {
        const int total = 4 * NHEAD * DQ * DIM_IN;   // 4 * 1048576
        convert_weights_kernel<<<total / 256, 256, 0, stream>>>(
            Wq, Wk, Wv, Wo, WqT, WkT, WvT, WoT);
    }
    // 1) QKV projection (z: 0=q, 1=k, 2=v)
    qkv_proj_kernel<<<dim3(SEQ / 64, BATCH * NHEAD, 3), 128, 0, stream>>>(
        Q, K, V, WqT, WkT, WvT, bq, bk, bv, qhBuf, khBuf, vTBuf);
    // 2) flash attention
    attn_kernel<<<dim3(SEQ / 64, BATCH * NHEAD), 128, 0, stream>>>(
        qhBuf, khBuf, vTBuf, concat);
    // 3) output projection + bias -> f32
    out_proj_kernel<<<dim3(4, SEQ / 16, BATCH), 128, 0, stream>>>(
        concat, WoT, bo, out);
}
